// CELLI_29850022707545
// MI455X (gfx1250) — matrix operand order repaired and checked by reading, not compiled, not run on hardware
//
#include <hip/hip_runtime.h>
#include <hip/hip_bf16.h>
#include <math.h>

typedef __attribute__((ext_vector_type(16))) _Float16 v16h;
typedef __attribute__((ext_vector_type(8)))  _Float16 v8h;
typedef __attribute__((ext_vector_type(8)))  float    v8f;

#define WMMA_F16(A,B,C) __builtin_amdgcn_wmma_f32_16x16x32_f16(false,(A),false,(B),(short)0,(C),false,false)

// fast silu: x * v_rcp_f32(1 + exp(-x)) -- exp and rcp are TRANS ops (co-execute with WMMA)
__device__ __forceinline__ float siluf(float v) {
    return v * __builtin_amdgcn_rcpf(1.f + __expf(-v));
}
__device__ __forceinline__ float softplusf(float v) { return (v > 20.f) ? v : log1pf(expf(v)); }
__device__ __forceinline__ float envelopef(float r) {
    // p=6: 1 - 28 r^6 + 48 r^7 - 21 r^8, zero for r >= 1
    if (r >= 1.f) return 0.f;
    float r2 = r*r, r3 = r2*r, r6 = r3*r3;
    return 1.f + r6*(-28.f + r*(48.f - 21.f*r));
}

// ---------------- zero scratch + pot ----------------
__global__ void k_zero(float* __restrict__ nodechi, long long N, float* __restrict__ pot) {
    long long i = (long long)blockIdx.x*256 + threadIdx.x;
    if (i < N) nodechi[i] = 0.f;
    if (i == 0) pot[0] = 0.f;
}

// ---------------- V passthrough copy ----------------
__global__ void k_copy(const float* __restrict__ src, float* __restrict__ dst, long long n) {
    long long i = (long long)blockIdx.x*256 + threadIdx.x;
    if (i < n) dst[i] = src[i];
}

// ---------------- per-edge chi MLP (32->16->1) + envelope + scatter-add ----------------
__global__ void __launch_bounds__(256) k_edge_chi(
    const float* __restrict__ vec, const float* __restrict__ x,
    const int* __restrict__ senders,
    const float* __restrict__ Wc1, const float* __restrict__ Wc2,
    float* __restrict__ env_ws, float* __restrict__ nodechi, long long E)
{
    __shared__ float w1s[32*16];
    __shared__ float w2s[16];
    const int t = threadIdx.x;
    const float rs32 = 0.17677669529663687f;  // 1/sqrt(32)
    const float rs16 = 0.25f;                 // 1/sqrt(16)
    for (int i = t; i < 512; i += 256) w1s[i] = Wc1[i] * rs32;
    if (t < 16) w2s[t] = Wc2[t] * rs16;
    __syncthreads();

    long long e = (long long)blockIdx.x*256 + t;
    if (e >= E) return;

    float vx = vec[e*3+0], vy = vec[e*3+1], vz = vec[e*3+2];
    float r  = sqrtf(vx*vx + vy*vy + vz*vz);
    float ev = envelopef(r);
    env_ws[e] = ev;

    const float* xe = x + e*32;
    float h[16];
    #pragma unroll
    for (int j = 0; j < 16; ++j) h[j] = 0.f;
    #pragma unroll
    for (int k = 0; k < 32; ++k) {
        float xv = xe[k];
        #pragma unroll
        for (int j = 0; j < 16; ++j) h[j] += xv * w1s[k*16 + j];
    }
    float chi = 0.f;
    #pragma unroll
    for (int j = 0; j < 16; ++j) chi += siluf(h[j]) * w2s[j];
    chi *= ev;
    atomicAdd(&nodechi[senders[e]], chi);
}

// ---------------- per-node: charges, pot, w-embedding (17->16, no act) ----------------
__global__ void __launch_bounds__(256) k_node(
    const float* __restrict__ nodechi, const int* __restrict__ species,
    const float* __restrict__ radius, const float* __restrict__ hardness,
    const float* __restrict__ emb, const float* __restrict__ scale,
    const float* __restrict__ shift, const float* __restrict__ W_w,
    float* __restrict__ charges_out, float* __restrict__ pot_out,
    _Float16* __restrict__ w16, long long N)
{
    __shared__ float wws[17*16];
    __shared__ float red[256];
    const int t = threadIdx.x;
    const float rs17 = 0.2425356250363330f;   // 1/sqrt(17)
    for (int i = t; i < 272; i += 256) wws[i] = W_w[i] * rs17;
    __syncthreads();

    long long n = (long long)blockIdx.x*256 + t;
    float potterm = 0.f;
    if (n < N) {
        float chi = nodechi[n] * scale[0] + shift[0];
        int sp = species[n];
        float hard = softplusf(hardness[sp]);
        float gam  = softplusf(radius[sp]) * 1.4426950408889634f;  // /ln(2)
        float q = -chi * __builtin_amdgcn_rcpf(hard);
        potterm = chi*q + 0.5f*hard*q*q + 0.01f*gam*q*q;
        charges_out[n] = q;
        float acc[16];
        #pragma unroll
        for (int j = 0; j < 16; ++j) acc[j] = q * wws[j];          // row 0: charge
        const float* em = emb + (long long)sp*16;
        #pragma unroll
        for (int i = 0; i < 16; ++i) {
            float evv = em[i];
            #pragma unroll
            for (int j = 0; j < 16; ++j) acc[j] += evv * wws[(i+1)*16 + j];
        }
        #pragma unroll
        for (int j = 0; j < 16; ++j) w16[n*16 + j] = (_Float16)acc[j];
    }
    red[t] = potterm;
    __syncthreads();
    for (int s = 128; s > 0; s >>= 1) {
        if (t < s) red[t] += red[t + s];
        __syncthreads();
    }
    if (t == 0) atomicAdd(pot_out, red[0]);
}

// ---------------- main per-edge MLP via WMMA: [E,48]@[48,32] -> [32,32] -> [32,32] ----------------
// xt/ht row stride = 56 halves (112B): 16B-aligned rows (b128 LDS ops) and
// conflict-free across the 16 lanes that vary in m (gcd(28 banks, 64) = 4, period 16).
__global__ void __launch_bounds__(256) k_edge_mlp(
    const float* __restrict__ x, const int* __restrict__ senders,
    const float* __restrict__ env, const _Float16* __restrict__ wnode,
    const float* __restrict__ W1, const float* __restrict__ W2, const float* __restrict__ W3,
    float* __restrict__ out, long long E)
{
    __shared__ _Float16 w1s[64][32];                       // rows 48..63 zero-padded
    __shared__ _Float16 w2s[32][32];
    __shared__ _Float16 w3s[32][32];
    __shared__ __align__(16) _Float16 xt[8][16][56];       // per-wave: x(0..31) + w[sender](32..47)
    __shared__ __align__(16) _Float16 ht[8][16][56];       // per-wave hidden activations
    __shared__ float    et[8][16];                         // per-wave envelope

    const int t = threadIdx.x;
    const float rs48 = 0.14433756729740643f;  // 1/sqrt(48)
    const float rs32 = 0.17677669529663687f;  // 1/sqrt(32)
    for (int i = t; i < 64*32; i += 256) {
        int r = i >> 5, c = i & 31;
        w1s[r][c] = (r < 48) ? (_Float16)(W1[r*32 + c] * rs48) : (_Float16)0.f;
    }
    for (int i = t; i < 32*32; i += 256) {
        int r = i >> 5, c = i & 31;
        w2s[r][c] = (_Float16)(W2[i] * rs32);
        w3s[r][c] = (_Float16)(W3[i] * rs32);
    }
    __syncthreads();

    const int lane = t & 31;
    const int wid  = t >> 5;
    const int nn   = lane & 15;   // N index (B/C/D), also M index for A reads
    const int hi   = lane >> 4;   // lane-half select

    // Persistent weight B-fragments (B layout: lane=N; VGPR v packs K = hi*8 + 2v + (v>=4 ? 8 : 0), +1)
    v16h b1[2][2], b2[2], b3[2];
    #pragma unroll
    for (int nt = 0; nt < 2; ++nt) {
        #pragma unroll
        for (int kt = 0; kt < 2; ++kt) {
            v16h b{};
            #pragma unroll
            for (int v = 0; v < 8; ++v) {
                int k = kt*32 + hi*8 + 2*v + (v >= 4 ? 8 : 0);
                b[2*v]   = w1s[k][nt*16 + nn];
                b[2*v+1] = w1s[k+1][nt*16 + nn];
            }
            b1[kt][nt] = b;
        }
        v16h c2{}, c3{};
        #pragma unroll
        for (int v = 0; v < 8; ++v) {
            int k = hi*8 + 2*v + (v >= 4 ? 8 : 0);
            c2[2*v] = w2s[k][nt*16 + nn]; c2[2*v+1] = w2s[k+1][nt*16 + nn];
            c3[2*v] = w3s[k][nt*16 + nn]; c3[2*v+1] = w3s[k+1][nt*16 + nn];
        }
        b2[nt] = c2; b3[nt] = c3;
    }

    const long long ntiles = (E + 15) >> 4;
    const long long nwaves = (long long)gridDim.x * 8;
    const long long Em1 = E - 1;

    for (long long tile = (long long)blockIdx.x*8 + wid; tile < ntiles; tile += nwaves) {
        const long long base = tile << 4;
        const bool full = (base + 16 <= E);

        // prefetch next tile's streaming inputs (global_prefetch_b8, per-lane cachelines)
        {
            long long ntile = tile + nwaves;
            if (ntile < ntiles) {
                const char* pfx = (const char*)(x + (ntile << 4) * 32);
                __builtin_prefetch(pfx + lane * 64, 0, 3);       // 32 lanes x 64B = 2KB tile
                __builtin_prefetch(senders + (ntile << 4), 0, 3);
                __builtin_prefetch(env + (ntile << 4), 0, 3);
            }
        }

        // stage x tile (16x32 f32 -> f16 LDS): 2 lanes per row, 4x float4 loads + 2x b128 LDS stores
        {
            int r = lane >> 1, c = (lane & 1) * 16;
            long long ee = base + r;
            if (ee > Em1) ee = Em1;               // tail rows: clamped garbage, stores are guarded
            const float4* xp = (const float4*)(x + ee*32 + c);
            float4 f0 = xp[0], f1 = xp[1], f2 = xp[2], f3 = xp[3];
            v8h h0, h1;
            h0[0]=(_Float16)f0.x; h0[1]=(_Float16)f0.y; h0[2]=(_Float16)f0.z; h0[3]=(_Float16)f0.w;
            h0[4]=(_Float16)f1.x; h0[5]=(_Float16)f1.y; h0[6]=(_Float16)f1.z; h0[7]=(_Float16)f1.w;
            h1[0]=(_Float16)f2.x; h1[1]=(_Float16)f2.y; h1[2]=(_Float16)f2.z; h1[3]=(_Float16)f2.w;
            h1[4]=(_Float16)f3.x; h1[5]=(_Float16)f3.y; h1[6]=(_Float16)f3.z; h1[7]=(_Float16)f3.w;
            *(v8h*)&xt[wid][r][c]     = h0;
            *(v8h*)&xt[wid][r][c + 8] = h1;
            if ((lane & 1) == 0) et[wid][r] = env[ee];
        }
        // stage gathered node embedding: one b128 global load + one b128 LDS store per lane
        {
            int er = lane & 15;
            long long ee = base + er;
            if (ee > Em1) ee = Em1;
            int s = senders[ee];
            *(v8h*)&xt[wid][er][32 + hi*8] = *(const v8h*)(wnode + (long long)s*16 + hi*8);
        }
        asm volatile("s_wait_dscnt 0" ::: "memory");

        // A fragments: per lane, two contiguous 16B runs -> 2x ds_load_b128 each
        v16h a0, a1{};
        {
            v8h lo = *(const v8h*)&xt[wid][nn][hi*8];
            v8h hp = *(const v8h*)&xt[wid][nn][16 + hi*8];
            #pragma unroll
            for (int i = 0; i < 8; ++i) { a0[i] = lo[i]; a0[8+i] = hp[i]; }
            v8h wlo = *(const v8h*)&xt[wid][nn][32 + hi*8];
            #pragma unroll
            for (int i = 0; i < 8; ++i) a1[i] = wlo[i];    // K 48..63 stays zero
        }

        // layer 1: K=48 (two K=32 WMMAs, second zero-padded), two N-tiles
        v8f c0{}, c1{};
        c0 = WMMA_F16(a0, b1[0][0], c0);
        c0 = WMMA_F16(a1, b1[1][0], c0);
        c1 = WMMA_F16(a0, b1[0][1], c1);
        c1 = WMMA_F16(a1, b1[1][1], c1);
        #pragma unroll
        for (int r = 0; r < 8; ++r) {          // D layout -> LDS [m][n]
            ht[wid][r + 8*hi][nn]      = (_Float16)siluf(c0[r]);
            ht[wid][r + 8*hi][16 + nn] = (_Float16)siluf(c1[r]);
        }
        asm volatile("s_wait_dscnt 0" ::: "memory");

        v16h a2;
        {
            v8h lo = *(const v8h*)&ht[wid][nn][hi*8];
            v8h hp = *(const v8h*)&ht[wid][nn][16 + hi*8];
            #pragma unroll
            for (int i = 0; i < 8; ++i) { a2[i] = lo[i]; a2[8+i] = hp[i]; }
        }
        // layer 2: one K=32 WMMA per N-tile
        v8f d0{}, d1{};
        d0 = WMMA_F16(a2, b2[0], d0);
        d1 = WMMA_F16(a2, b2[1], d1);
        asm volatile("s_wait_dscnt 0" ::: "memory");
        #pragma unroll
        for (int r = 0; r < 8; ++r) {
            ht[wid][r + 8*hi][nn]      = (_Float16)siluf(d0[r]);
            ht[wid][r + 8*hi][16 + nn] = (_Float16)siluf(d1[r]);
        }
        asm volatile("s_wait_dscnt 0" ::: "memory");

        v16h a3;
        {
            v8h lo = *(const v8h*)&ht[wid][nn][hi*8];
            v8h hp = *(const v8h*)&ht[wid][nn][16 + hi*8];
            #pragma unroll
            for (int i = 0; i < 8; ++i) { a3[i] = lo[i]; a3[8+i] = hp[i]; }
        }
        // layer 3 (no activation) + envelope scale + store
        v8f f0{}, f1{};
        f0 = WMMA_F16(a3, b3[0], f0);
        f1 = WMMA_F16(a3, b3[1], f1);
        if (full) {
            #pragma unroll
            for (int r = 0; r < 8; ++r) {
                int mm = r + 8*hi;
                long long ee = base + mm;
                float evv = et[wid][mm];
                out[ee*32 + nn]      = f0[r] * evv;
                out[ee*32 + 16 + nn] = f1[r] * evv;
            }
        } else {
            #pragma unroll
            for (int r = 0; r < 8; ++r) {
                int mm = r + 8*hi;
                long long ee = base + mm;
                if (ee < E) {
                    float evv = et[wid][mm];
                    out[ee*32 + nn]      = f0[r] * evv;
                    out[ee*32 + 16 + nn] = f1[r] * evv;
                }
            }
        }
    }
}

extern "C" void kernel_launch(void* const* d_in, const int* in_sizes, int n_in,
                              void* d_out, int out_size, void* d_ws, size_t ws_size,
                              hipStream_t stream) {
    const float* vectors  = (const float*)d_in[0];
    const float* x        = (const float*)d_in[1];
    const float* V        = (const float*)d_in[2];
    const int*   senders  = (const int*)  d_in[3];
    const int*   species  = (const int*)  d_in[4];
    const float* W_chi1   = (const float*)d_in[5];
    const float* W_chi2   = (const float*)d_in[6];
    const float* radius   = (const float*)d_in[7];
    const float* hardness = (const float*)d_in[8];
    const float* emb      = (const float*)d_in[9];
    const float* scale    = (const float*)d_in[10];
    const float* shift    = (const float*)d_in[11];
    const float* W_w      = (const float*)d_in[12];
    const float* W_x1     = (const float*)d_in[13];
    const float* W_x2     = (const float*)d_in[14];
    const float* W_x3     = (const float*)d_in[15];

    const long long E = in_sizes[0] / 3;     // edges
    const long long N = in_sizes[2] / 16;    // nodes (V is [N,16])

    // d_out layout: out[E,32] | V[N,16] | charges[N] | pot[1]
    float* out     = (float*)d_out;
    float* vout    = out + E*32;
    float* charges = vout + N*16;
    float* pot     = charges + N;

    // workspace: env[E] f32 | nodechi[N] f32 | w16[N,16] f16   (~8.2 MB)
    float*    env_ws  = (float*)d_ws;
    float*    nodechi = env_ws + E;
    _Float16* w16     = (_Float16*)(nodechi + N);

    {
        long long blk = (N + 255) / 256;
        k_zero<<<(unsigned)blk, 256, 0, stream>>>(nodechi, N, pot);
    }
    {
        long long blk = (E + 255) / 256;
        k_edge_chi<<<(unsigned)blk, 256, 0, stream>>>(vectors, x, senders, W_chi1, W_chi2,
                                                      env_ws, nodechi, E);
    }
    {
        long long blk = (N*16 + 255) / 256;
        k_copy<<<(unsigned)blk, 256, 0, stream>>>(V, vout, N*16);
    }
    {
        long long blk = (N + 255) / 256;
        k_node<<<(unsigned)blk, 256, 0, stream>>>(nodechi, species, radius, hardness, emb,
                                                  scale, shift, W_w, charges, pot, w16, N);
    }
    {
        long long ntiles = (E + 15) / 16;
        long long blk = (ntiles + 7) / 8;
        if (blk > 2048) blk = 2048;
        k_edge_mlp<<<(unsigned)blk, 256, 0, stream>>>(x, senders, env_ws, w16,
                                                      W_x1, W_x2, W_x3, out, E);
    }
}
